// Transformer_P_80204219286216
// MI455X (gfx1250) — matrix operand order repaired and checked by reading, not compiled, not run on hardware
//
#include <hip/hip_runtime.h>
#include <math.h>

typedef __attribute__((ext_vector_type(16))) _Float16 v16h;
typedef __attribute__((ext_vector_type(8)))  float    v8f;
typedef __attribute__((ext_vector_type(4)))  float    f4;
typedef __attribute__((ext_vector_type(4)))  _Float16 h4;

// ---------------- problem constants ----------------
static constexpr int BSZ   = 4;
static constexpr int N0    = 6000;
static constexpr int NT    = 6085;     // 1 + 78*78
static constexpr int BN    = BSZ * NT; // 24340
static constexpr int D     = 512;
static constexpr int HN    = 8;
static constexpr int DH    = 64;
static constexpr int MLAND = 256;
static constexpr int NPD   = 6144;     // padded seq per batch
static constexpr int PAD   = 59;       // front pad
static constexpr int LSEG  = 24;       // NPD / MLAND
static constexpr int CONVK = 33;

// ---------------- WMMA GEMM ----------------
// C[bz] = alpha * (A[bz] @ B[bz](^T)) + diag*I + bias[bz]
// A: f32 [M,K] row-major, B: f32 [K,N] (or [N,K] if transB), C: f32 [M,N]
static constexpr int TB_M = 128, TB_N = 128, TB_K = 32;

__device__ __forceinline__ void stage_tile(
    _Float16 (*__restrict__ As)[TB_K], _Float16 (*__restrict__ Bs)[TB_K],
    const float* __restrict__ Ab, const float* __restrict__ Bb,
    int M, int N, int K, int row0, int col0, int k0, int transB, int tid)
{
    const bool kFull = (k0 + TB_K <= K);
    // ---- A tile: rows contiguous along K ----
    if ((row0 + TB_M <= M) && kFull) {
#pragma unroll
        for (int it = 0; it < 4; ++it) {
            int i = tid + it * 256;
            int r = i >> 3, kq = (i & 7) * 4;
            f4 v = *(const f4*)(Ab + (long long)(row0 + r) * K + k0 + kq);
            h4 hv; hv[0] = (_Float16)v[0]; hv[1] = (_Float16)v[1];
                   hv[2] = (_Float16)v[2]; hv[3] = (_Float16)v[3];
            *(h4*)(&As[r][kq]) = hv;
        }
    } else {
        for (int i = tid; i < TB_M * TB_K; i += 256) {
            int r = i >> 5, kk = i & 31;
            int gr = row0 + r, gk = k0 + kk;
            float v = (gr < M && gk < K) ? Ab[(long long)gr * K + gk] : 0.0f;
            As[r][kk] = (_Float16)v;
        }
    }
    // ---- B tile: stored as Bs[n][k] ----
    if (transB) {
        if ((col0 + TB_N <= N) && kFull) {
#pragma unroll
            for (int it = 0; it < 4; ++it) {
                int i = tid + it * 256;
                int c = i >> 3, kq = (i & 7) * 4;
                f4 v = *(const f4*)(Bb + (long long)(col0 + c) * K + k0 + kq);
                h4 hv; hv[0] = (_Float16)v[0]; hv[1] = (_Float16)v[1];
                       hv[2] = (_Float16)v[2]; hv[3] = (_Float16)v[3];
                *(h4*)(&Bs[c][kq]) = hv;
            }
        } else {
            for (int i = tid; i < TB_N * TB_K; i += 256) {
                int c = i >> 5, kk = i & 31;
                int gc = col0 + c, gk = k0 + kk;
                float v = (gc < N && gk < K) ? Bb[(long long)gc * K + gk] : 0.0f;
                Bs[c][kk] = (_Float16)v;
            }
        }
    } else {
        if ((col0 + TB_N <= N) && kFull) {
#pragma unroll
            for (int it = 0; it < 4; ++it) {
                int i = tid + it * 256;
                int kk = i >> 5, nq = (i & 31) * 4;
                f4 v = *(const f4*)(Bb + (long long)(k0 + kk) * N + col0 + nq);
#pragma unroll
                for (int j = 0; j < 4; ++j) Bs[nq + j][kk] = (_Float16)v[j];
            }
        } else {
            for (int i = tid; i < TB_N * TB_K; i += 256) {
                int c = i >> 5, kk = i & 31;
                int gc = col0 + c, gk = k0 + kk;
                float v = (gc < N && gk < K) ? Bb[(long long)gk * N + gc] : 0.0f;
                Bs[c][kk] = (_Float16)v;
            }
        }
    }
}

__global__ __launch_bounds__(256)
void gemm_wmma(const float* __restrict__ A, const float* __restrict__ Bm,
               float* __restrict__ C,
               int M, int N, int K,
               long long sA, long long sB, long long sC,
               int transB, float alpha, float diag,
               const float* __restrict__ bias, long long sBias)
{
    __shared__ _Float16 As[2][TB_M][TB_K];
    __shared__ _Float16 Bs[2][TB_N][TB_K];

    const int bz = blockIdx.z;
    const float* Ab = A  + (long long)bz * sA;
    const float* Bb = Bm + (long long)bz * sB;
    float*       Cb = C  + (long long)bz * sC;
    const float* biasb = bias ? bias + (long long)bz * sBias : nullptr;

    const int row0 = blockIdx.y * TB_M;
    const int col0 = blockIdx.x * TB_N;

    const int tid   = threadIdx.x;
    const int wid   = tid >> 5;
    const int lane  = tid & 31;
    const int wm    = wid & 3;   // 4 waves along M (32 rows each)
    const int wn    = wid >> 2;  // 2 waves along N (64 cols each)
    const int hi    = lane >> 4;
    const int lanem = lane & 15;

    v8f acc[2][4];
    for (int i = 0; i < 2; ++i)
        for (int j = 0; j < 4; ++j)
            for (int e = 0; e < 8; ++e) acc[i][j][e] = 0.0f;

    const int nsteps = (K + TB_K - 1) / TB_K;
    stage_tile(As[0], Bs[0], Ab, Bb, M, N, K, row0, col0, 0, transB, tid);
    __syncthreads();

    for (int s = 0; s < nsteps; ++s) {
        const int k0  = s * TB_K;
        const int cur = s & 1;

        if (s + 2 < nsteps) {   // pull tile s+2 toward cache (global_prefetch_b8)
            int kpf = k0 + 2 * TB_K;
            __builtin_prefetch(Ab + (long long)(row0 + (tid >> 1)) * K + kpf, 0, 1);
            __builtin_prefetch(transB
                ? Bb + (long long)(col0 + (tid >> 1)) * K + kpf
                : Bb + (long long)kpf * N + col0 + (tid >> 1) * 2, 0, 1);
        }
        if (s + 1 < nsteps)     // stage next tile into the other LDS buffer
            stage_tile(As[cur ^ 1], Bs[cur ^ 1], Ab, Bb, M, N, K,
                       row0, col0, k0 + TB_K, transB, tid);

        // A frags: lane m = lane&15; K halves {hi*8+j, 16+hi*8+j} per ISA 16-bit A layout
        v16h afrag[2];
        for (int tm = 0; tm < 2; ++tm) {
            int r = wm * 32 + tm * 16 + lanem;
            const _Float16* p0 = &As[cur][r][hi * 8];
            const _Float16* p1 = &As[cur][r][16 + hi * 8];
            for (int j = 0; j < 8; ++j) afrag[tm][j]     = p0[j];
            for (int j = 0; j < 8; ++j) afrag[tm][8 + j] = p1[j];
        }
        // B frags: n = lane&15, K halves {hi*8+j, 16+hi*8+j} (matching A)
        v16h bfrag[4];
        for (int tn = 0; tn < 4; ++tn) {
            int c = wn * 64 + tn * 16 + lanem;
            const _Float16* p0 = &Bs[cur][c][hi * 8];
            const _Float16* p1 = &Bs[cur][c][16 + hi * 8];
            for (int j = 0; j < 8; ++j) bfrag[tn][j]     = p0[j];
            for (int j = 0; j < 8; ++j) bfrag[tn][8 + j] = p1[j];
        }
        for (int tm = 0; tm < 2; ++tm)
            for (int tn = 0; tn < 4; ++tn)
                acc[tm][tn] = __builtin_amdgcn_wmma_f32_16x16x32_f16(
                    false, afrag[tm], false, bfrag[tn],
                    (short)0, acc[tm][tn], false, false);
        __syncthreads();
    }

    // epilogue: C/D layout VGPR r -> M=r (lanes0-15) / r+8 (lanes16-31), N=lane&15
    const bool full = (row0 + TB_M <= M) && (col0 + TB_N <= N);
    if (full) {
        for (int tm = 0; tm < 2; ++tm)
            for (int tn = 0; tn < 4; ++tn)
                for (int r = 0; r < 8; ++r) {
                    int gm = row0 + wm * 32 + tm * 16 + r + hi * 8;
                    int gn = col0 + wn * 64 + tn * 16 + lanem;
                    float v = alpha * acc[tm][tn][r];
                    if (gm == gn) v += diag;
                    if (biasb) v += biasb[gn];
                    Cb[(long long)gm * N + gn] = v;
                }
    } else {
        for (int tm = 0; tm < 2; ++tm)
            for (int tn = 0; tn < 4; ++tn)
                for (int r = 0; r < 8; ++r) {
                    int gm = row0 + wm * 32 + tm * 16 + r + hi * 8;
                    int gn = col0 + wn * 64 + tn * 16 + lanem;
                    if (gm < M && gn < N) {
                        float v = alpha * acc[tm][tn][r];
                        if (gm == gn) v += diag;
                        if (biasb) v += biasb[gn];
                        Cb[(long long)gm * N + gn] = v;
                    }
                }
    }
}

// ---------------- elementwise / reduction kernels ----------------
__global__ void fill_f32(float* p, long long n, float v) {
    long long i = (long long)blockIdx.x * blockDim.x + threadIdx.x;
    if (i < n) p[i] = v;
}

__global__ void build_input(const float* __restrict__ feat,
                            const float* __restrict__ cls,
                            float* __restrict__ h, long long n) {
    long long i = (long long)blockIdx.x * blockDim.x + threadIdx.x;
    if (i >= n) return;
    int d = (int)(i % D);
    long long row = i / D;
    int t = (int)(row % NT), b = (int)(row / NT);
    float v;
    if (t == 0) v = cls[d];
    else {
        int s = t - 1;
        if (s >= N0) s -= N0;   // wrap pad
        v = feat[((long long)b * N0 + s) * D + d];
    }
    h[i] = v;
}

// LN over D=512; orow = (row/inPerB)*outPerB + pad + row%inPerB ; gamma/beta per (row/gbRowsPer)
__global__ __launch_bounds__(256)
void ln_kernel(const float* __restrict__ in, float* __restrict__ out,
               const float* __restrict__ g, const float* __restrict__ b,
               long long rows, long long inPerB, long long outPerB, int pad,
               long long gbRowsPer, int relu)
{
    long long row = blockIdx.x;
    if (row >= rows) return;
    const float* x = in + row * D;
    long long orow = (row / inPerB) * outPerB + pad + (row % inPerB);
    __shared__ float red[256];
    int tid = threadIdx.x;
    float s = 0.f;
    for (int d2 = tid; d2 < D; d2 += 256) s += x[d2];
    red[tid] = s; __syncthreads();
    for (int o = 128; o > 0; o >>= 1) { if (tid < o) red[tid] += red[tid + o]; __syncthreads(); }
    float mu = red[0] * (1.f / D);
    __syncthreads();
    float vs = 0.f;
    for (int d2 = tid; d2 < D; d2 += 256) { float t = x[d2] - mu; vs += t * t; }
    red[tid] = vs; __syncthreads();
    for (int o = 128; o > 0; o >>= 1) { if (tid < o) red[tid] += red[tid + o]; __syncthreads(); }
    float inv = rsqrtf(red[0] * (1.f / D) + 1e-5f);
    long long gi = gbRowsPer > 0 ? (row / gbRowsPer) : 0;
    const float* gp = g + gi * D;
    const float* bp = b + gi * D;
    float* op = out + orow * D;
    for (int d2 = tid; d2 < D; d2 += 256) {
        float y = (x[d2] - mu) * inv * gp[d2] + bp[d2];
        if (relu) y = fmaxf(y, 0.f);
        op[d2] = y;
    }
}

__global__ __launch_bounds__(256)
void softmax_rows(float* __restrict__ buf, long long rows, int width) {
    long long row = blockIdx.x;
    if (row >= rows) return;
    float* x = buf + row * (long long)width;
    __shared__ float red[256];
    int tid = threadIdx.x;
    float m = -1e30f;
    for (int i = tid; i < width; i += 256) m = fmaxf(m, x[i]);
    red[tid] = m; __syncthreads();
    for (int o = 128; o > 0; o >>= 1) { if (tid < o) red[tid] = fmaxf(red[tid], red[tid + o]); __syncthreads(); }
    float M = red[0];
    __syncthreads();
    float s = 0.f;
    for (int i = tid; i < width; i += 256) { float e = __expf(x[i] - M); x[i] = e; s += e; }
    red[tid] = s; __syncthreads();
    for (int o = 128; o > 0; o >>= 1) { if (tid < o) red[tid] += red[tid + o]; __syncthreads(); }
    float inv = 1.f / red[0];
    for (int i = tid; i < width; i += 256) x[i] *= inv;
}

__global__ void moe_gate(const float* __restrict__ x, const float* __restrict__ gw,
                         const float* __restrict__ gb, int E, int k,
                         int* __restrict__ idx, float* __restrict__ wts) {
    int t = blockIdx.x * blockDim.x + threadIdx.x;
    if (t >= BN) return;
    float s[4];
    for (int e = 0; e < E; ++e) s[e] = gb[e];
    const float* xp = x + (long long)t * D;
    for (int d2 = 0; d2 < D; ++d2) {
        float xv = xp[d2];
        for (int e = 0; e < E; ++e) s[e] += xv * gw[d2 * E + e];
    }
    int i0 = 0;
    for (int e = 1; e < E; ++e) if (s[e] > s[i0]) i0 = e;
    if (k == 1) { idx[t] = i0; wts[t] = 1.f; return; }
    int i1 = -1;
    for (int e = 0; e < E; ++e) { if (e == i0) continue; if (i1 < 0 || s[e] > s[i1]) i1 = e; }
    float b2 = __expf(s[i1] - s[i0]);
    float inv = 1.f / (1.f + b2);
    idx[2 * t] = i0; idx[2 * t + 1] = i1;
    wts[2 * t] = inv; wts[2 * t + 1] = b2 * inv;
}

__global__ void moe_combine(float* __restrict__ h, const float* __restrict__ obuf,
                            const int* __restrict__ idx, const float* __restrict__ wts,
                            int k, long long n) {
    long long i = (long long)blockIdx.x * blockDim.x + threadIdx.x;
    if (i >= n) return;
    int d = (int)(i % D);
    long long t = i / D;
    float s = 0.f;
    for (int j = 0; j < k; ++j) {
        int e = idx[t * k + j];
        s += wts[t * k + j] * obuf[(long long)e * BN * D + t * D + d];
    }
    h[i] += s;
}

__global__ void split_qkv(const float* __restrict__ qkv, float* __restrict__ q,
                          float* __restrict__ kk, float* __restrict__ v, long long n) {
    long long idx = (long long)blockIdx.x * blockDim.x + threadIdx.x;
    if (idx >= n) return;
    int d512 = (int)(idx % D);
    long long bi = idx / D;
    int i = (int)(bi % NPD), b = (int)(bi / NPD);
    int hh = d512 >> 6, dh = d512 & 63;
    const float* src = qkv + ((long long)b * NPD + i) * (3 * D);
    long long o = (((long long)(b * HN + hh)) * NPD + i) * DH + dh;
    q[o]  = src[hh * DH + dh] * 0.125f;   // dh^-0.5
    kk[o] = src[D + hh * DH + dh];
    v[o]  = src[2 * D + hh * DH + dh];
}

__global__ void landmarks(const float* __restrict__ q, float* __restrict__ ql, long long n) {
    long long idx = (long long)blockIdx.x * blockDim.x + threadIdx.x;
    if (idx >= n) return;
    int dh = (int)(idx & 63);
    int mm = (int)((idx >> 6) & 255);
    long long bh = idx >> 14;
    const float* p = q + ((long long)bh * NPD + (long long)mm * LSEG) * DH + dh;
    float s = 0.f;
    for (int l = 0; l < LSEG; ++l) s += p[(long long)l * DH];
    ql[idx] = s * (1.f / LSEG);
}

__global__ __launch_bounds__(256)
void pinv_norms(const float* __restrict__ x, float* __restrict__ scal) {
    const float* xb = x + (long long)blockIdx.x * MLAND * MLAND;
    int tid = threadIdx.x;
    float rs = 0.f, cs = 0.f;
    for (int j = 0; j < MLAND; ++j) {
        rs += fabsf(xb[tid * MLAND + j]);
        cs += fabsf(xb[j * MLAND + tid]);
    }
    __shared__ float r1[256], r2[256];
    r1[tid] = rs; r2[tid] = cs; __syncthreads();
    for (int o = 128; o > 0; o >>= 1) {
        if (tid < o) { r1[tid] = fmaxf(r1[tid], r1[tid + o]); r2[tid] = fmaxf(r2[tid], r2[tid + o]); }
        __syncthreads();
    }
    if (tid == 0) {
        atomicMax((unsigned int*)&scal[0], __float_as_uint(r1[0]));
        atomicMax((unsigned int*)&scal[1], __float_as_uint(r2[0]));
    }
}

__global__ void pinv_init(const float* __restrict__ x, const float* __restrict__ scal,
                          float* __restrict__ z, long long n) {
    long long idx = (long long)blockIdx.x * blockDim.x + threadIdx.x;
    if (idx >= n) return;
    int j = (int)(idx % MLAND);
    int i = (int)((idx / MLAND) % MLAND);
    long long bh = idx / ((long long)MLAND * MLAND);
    float s = 1.f / (scal[0] * scal[1]);
    z[idx] = x[bh * MLAND * MLAND + (long long)j * MLAND + i] * s;   // transpose
}

__global__ void diag_sub(const float* __restrict__ in, float* __restrict__ out,
                         float diag, long long n) {
    long long idx = (long long)blockIdx.x * blockDim.x + threadIdx.x;
    if (idx >= n) return;
    int j = (int)(idx % MLAND);
    int i = (int)((idx / MLAND) % MLAND);
    out[idx] = (i == j ? diag : 0.f) - in[idx];
}

__global__ void conv_res_add(const float* __restrict__ v, const float* __restrict__ w,
                             float* __restrict__ o, long long n) {
    long long idx = (long long)blockIdx.x * blockDim.x + threadIdx.x;
    if (idx >= n) return;
    int d = (int)(idx & 63);
    long long r = idx >> 6;
    int i = (int)(r % NPD);
    long long bh = r / NPD;
    int hh = (int)(bh & 7);
    const float* vb = v + bh * NPD * DH + d;
    float s = 0.f;
    for (int k2 = 0; k2 < CONVK; ++k2) {
        int ii = i + k2 - CONVK / 2;
        if (ii >= 0 && ii < NPD) s += w[hh * CONVK + k2] * vb[(long long)ii * DH];
    }
    o[idx] += s;
}

__global__ void merge_heads(const float* __restrict__ o, float* __restrict__ ob, long long n) {
    long long idx = (long long)blockIdx.x * blockDim.x + threadIdx.x;
    if (idx >= n) return;
    int d512 = (int)(idx % D);
    long long bi = idx / D;
    int i = (int)(bi % NPD), b = (int)(bi / NPD);
    int hh = d512 >> 6, d = d512 & 63;
    ob[idx] = o[(((long long)(b * HN + hh)) * NPD + i) * DH + d];
}

__global__ void add_unpad(float* __restrict__ h, const float* __restrict__ ao, long long n) {
    long long idx = (long long)blockIdx.x * blockDim.x + threadIdx.x;
    if (idx >= n) return;
    int d = (int)(idx % D);
    long long r = idx / D;
    int t = (int)(r % NT), b = (int)(r / NT);
    h[idx] += ao[((long long)b * NPD + PAD + t) * D + d];
}

__global__ void write_out(const float* __restrict__ h, float* __restrict__ out, long long n) {
    long long idx = (long long)blockIdx.x * blockDim.x + threadIdx.x;
    if (idx >= n) return;
    int d = (int)(idx % D);
    long long r = idx / D;
    int t = (int)(r % NT), b = (int)(r / NT);
    float v = h[idx];
    if (t == 0) out[(long long)b * D + d] = v;
    else out[(long long)BSZ * D + ((long long)b * (NT - 1) + (t - 1)) * D + d] = v;
}

// ---------------- host orchestration ----------------
static inline dim3 g1(long long n) { return dim3((unsigned)((n + 255) / 256)); }

extern "C" void kernel_launch(void* const* d_in, const int* in_sizes, int n_in,
                              void* d_out, int out_size, void* d_ws, size_t ws_size,
                              hipStream_t stream) {
    const float* features = (const float*)d_in[0];
    const float* cls      = (const float*)d_in[1];
    const float* m1[10]; for (int i = 0; i < 10; ++i) m1[i] = (const float*)d_in[2 + i];
    const float* m2[10]; for (int i = 0; i < 10; ++i) m2[i] = (const float*)d_in[12 + i];
    const float* l1p[6]; for (int i = 0; i < 6;  ++i) l1p[i] = (const float*)d_in[22 + i];
    const float* l2p[6]; for (int i = 0; i < 6;  ++i) l2p[i] = (const float*)d_in[28 + i];

    // workspace bump allocator
    char* ws = (char*)d_ws;
    size_t off = 0;
    auto alloc = [&](size_t bytes) -> float* {
        float* p = (float*)(ws + off);
        off += (bytes + 255) & ~(size_t)255;
        return p;
    };
    const long long ND    = (long long)BN * D;                 // 12,462,080
    const long long NPDD  = (long long)BSZ * NPD * D;          // 12,582,912
    const long long TOTQ  = (long long)BSZ * HN * NPD * DH;    // 12,582,912
    const long long MM32  = 32LL * MLAND * MLAND;              // 2,097,152

    float* h    = alloc(ND * 4);
    float* R1   = alloc(32LL * NPD * MLAND * 4);   // attn1 / moe hidden (E<=4)
    float* R2   = alloc(32LL * NPD * MLAND * 4);   // attn3 / moe out / per-head out
    float* QKVB = alloc((long long)BSZ * NPD * 3 * D * 4);
    float* XPAD = alloc(NPDD * 4);
    float* Q    = alloc(TOTQ * 4);
    float* Kb   = alloc(TOTQ * 4);
    float* V    = alloc(TOTQ * 4);
    float* QL   = alloc(32LL * MLAND * DH * 4);
    float* KL   = alloc(32LL * MLAND * DH * 4);
    float* A2   = alloc(MM32 * 4);
    float* Z1   = alloc(MM32 * 4);
    float* Z2   = alloc(MM32 * 4);
    float* XZ   = alloc(MM32 * 4);
    float* TT   = alloc(MM32 * 4);
    float* UU   = alloc(MM32 * 4);
    float* TV   = alloc(32LL * MLAND * DH * 4);
    float* W2   = alloc(32LL * MLAND * DH * 4);
    int*   IDX  = (int*)alloc((long long)BN * 2 * 4);
    float* WTS  = alloc((long long)BN * 2 * 4);
    float* SCAL = alloc(256);
    float* OB   = QKVB;                 // alias: qkv dead after split
    float* AOUT = QKVB + NPDD;          // alias: disjoint from OB within qkv buffer
    float* OH   = R2;                   // alias: attn3 dead after attn3@V

    auto gemm = [&](const float* A, const float* Bm, float* C, int M, int N, int K,
                    long long sA, long long sB, long long sC, int batch, int transB,
                    float alpha, float diag, const float* bias, long long sBias) {
        dim3 grid((N + TB_N - 1) / TB_N, (M + TB_M - 1) / TB_M, batch);
        gemm_wmma<<<grid, dim3(256), 0, stream>>>(A, Bm, C, M, N, K, sA, sB, sC,
                                                  transB, alpha, diag, bias, sBias);
    };

    auto moe = [&](int E, int k, const float* const* p) {
        // p: gate_w, gate_b, fc1_w, fc1_b, ln1_g, ln1_b, fc2_w, fc2_b, ln2_g, ln2_b
        moe_gate<<<(BN + 127) / 128, 128, 0, stream>>>(h, p[0], p[1], E, k, IDX, WTS);
        gemm(h, p[2], R1, BN, D, D, 0, (long long)D * D, ND, E, 0, 1.f, 0.f, p[3], D);
        ln_kernel<<<(unsigned)((long long)E * BN), 256, 0, stream>>>(
            R1, R1, p[4], p[5], (long long)E * BN, BN, BN, 0, BN, 1);
        gemm(R1, p[6], R2, BN, D, D, ND, (long long)D * D, ND, E, 0, 1.f, 0.f, p[7], D);
        ln_kernel<<<(unsigned)((long long)E * BN), 256, 0, stream>>>(
            R2, R2, p[8], p[9], (long long)E * BN, BN, BN, 0, BN, 0);
        moe_combine<<<g1(ND), 256, 0, stream>>>(h, R2, IDX, WTS, k, ND);
    };

    auto attn = [&](const float* const* p) {
        // p: norm_g, norm_b, qkv_w, out_w, out_b, conv_w
        fill_f32<<<g1(NPDD), 256, 0, stream>>>(XPAD, NPDD, 0.f);
        ln_kernel<<<(unsigned)BN, 256, 0, stream>>>(h, XPAD, p[0], p[1],
                                                    BN, NT, NPD, PAD, 0, 0);
        gemm(XPAD, p[2], QKVB, BSZ * NPD, 3 * D, D, 0, 0, 0, 1, 0, 1.f, 0.f, nullptr, 0);
        split_qkv<<<g1(NPDD), 256, 0, stream>>>(QKVB, Q, Kb, V, NPDD);
        landmarks<<<g1(32LL * MLAND * DH), 256, 0, stream>>>(Q, QL, 32LL * MLAND * DH);
        landmarks<<<g1(32LL * MLAND * DH), 256, 0, stream>>>(Kb, KL, 32LL * MLAND * DH);
        // attn1 = softmax(q @ kl^T) : [32][NPD][256]
        gemm(Q, KL, R1, NPD, MLAND, DH, (long long)NPD * DH, (long long)MLAND * DH,
             (long long)NPD * MLAND, 32, 1, 1.f, 0.f, nullptr, 0);
        softmax_rows<<<(unsigned)(32LL * NPD), 256, 0, stream>>>(R1, 32LL * NPD, MLAND);
        // attn2 = softmax(ql @ kl^T) : [32][256][256]
        gemm(QL, KL, A2, MLAND, MLAND, DH, (long long)MLAND * DH, (long long)MLAND * DH,
             (long long)MLAND * MLAND, 32, 1, 1.f, 0.f, nullptr, 0);
        softmax_rows<<<32 * MLAND, 256, 0, stream>>>(A2, 32LL * MLAND, MLAND);
        // attn3 = softmax(ql @ k^T) : [32][256][NPD]
        gemm(QL, Kb, R2, MLAND, NPD, DH, (long long)MLAND * DH, (long long)NPD * DH,
             (long long)MLAND * NPD, 32, 1, 1.f, 0.f, nullptr, 0);
        softmax_rows<<<32 * MLAND, 256, 0, stream>>>(R2, 32LL * MLAND, NPD);
        // tv = attn3 @ v : [32][256][64]
        gemm(R2, V, TV, MLAND, DH, NPD, (long long)MLAND * NPD, (long long)NPD * DH,
             (long long)MLAND * DH, 32, 0, 1.f, 0.f, nullptr, 0);
        // Moore-Penrose pinv of attn2
        fill_f32<<<1, 256, 0, stream>>>(SCAL, 2, 0.f);
        pinv_norms<<<32, 256, 0, stream>>>(A2, SCAL);
        pinv_init<<<g1(MM32), 256, 0, stream>>>(A2, SCAL, Z1, MM32);
        float* zc = Z1; float* zn = Z2;
        const long long sM = (long long)MLAND * MLAND;
        for (int it = 0; it < 6; ++it) {
            gemm(A2, zc, XZ, MLAND, MLAND, MLAND, sM, sM, sM, 32, 0, 1.f,  0.f,  nullptr, 0);
            diag_sub<<<g1(MM32), 256, 0, stream>>>(XZ, TT, 7.f, MM32);
            gemm(XZ, TT, UU, MLAND, MLAND, MLAND, sM, sM, sM, 32, 0, -1.f, 15.f, nullptr, 0);
            gemm(XZ, UU, TT, MLAND, MLAND, MLAND, sM, sM, sM, 32, 0, -1.f, 13.f, nullptr, 0);
            gemm(zc, TT, zn, MLAND, MLAND, MLAND, sM, sM, sM, 32, 0, 0.25f, 0.f, nullptr, 0);
            float* t = zc; zc = zn; zn = t;
        }
        // w2 = z @ tv ; o = attn1 @ w2 (+ conv residual on v)
        gemm(zc, TV, W2, MLAND, DH, MLAND, sM, (long long)MLAND * DH,
             (long long)MLAND * DH, 32, 0, 1.f, 0.f, nullptr, 0);
        gemm(R1, W2, OH, NPD, DH, MLAND, (long long)NPD * MLAND, (long long)MLAND * DH,
             (long long)NPD * DH, 32, 0, 1.f, 0.f, nullptr, 0);
        conv_res_add<<<g1(TOTQ), 256, 0, stream>>>(V, p[5], OH, TOTQ);
        merge_heads<<<g1(NPDD), 256, 0, stream>>>(OH, OB, NPDD);
        gemm(OB, p[3], AOUT, BSZ * NPD, D, D, 0, 0, 0, 1, 0, 1.f, 0.f, p[4], 0);
        add_unpad<<<g1(ND), 256, 0, stream>>>(h, AOUT, ND);
    };

    build_input<<<g1(ND), 256, 0, stream>>>(features, cls, h, ND);
    moe(2, 1, m1);
    attn(l1p);
    moe(4, 2, m2);
    attn(l2p);
    write_out<<<g1(ND), 256, 0, stream>>>(h, (float*)d_out, ND);
}
